// MHLA_29841432772831
// MI455X (gfx1250) — matrix operand order repaired and checked by reading, not compiled, not run on hardware
//
#include <hip/hip_runtime.h>
#include <hip/hip_bf16.h>

// ---------- types ----------
typedef __attribute__((ext_vector_type(16))) __bf16       v16bf;
typedef __attribute__((ext_vector_type(8)))  float        v8f;
typedef __attribute__((ext_vector_type(4)))  unsigned int v4u;

#define D_MODEL 1024
#define N_HEADS 32
#define BATCH 2
#define SEQ 2048
#define MROWS (BATCH * SEQ)       // 4096
#define NQKV  (3 * D_MODEL)       // 3072

__device__ __forceinline__ unsigned short f2bf(float f) {
    union { float f; unsigned u; } v; v.f = f;
    unsigned r = v.u + 0x7FFFu + ((v.u >> 16) & 1u);   // round-to-nearest-even
    return (unsigned short)(r >> 16);
}

__device__ __forceinline__ v8f v8f_zero() {
    v8f z = {0.f,0.f,0.f,0.f,0.f,0.f,0.f,0.f};
    return z;
}

// load a 32-byte WMMA fragment from two 16-byte chunks
__device__ __forceinline__ v16bf ld_frag(const unsigned short* lo, const unsigned short* hi) {
    union { v4u u[2]; v16bf b; } t;
    t.u[0] = *reinterpret_cast<const v4u*>(lo);
    t.u[1] = *reinterpret_cast<const v4u*>(hi);
    return t.b;
}

// CDNA5 async global->LDS copy of one 16B chunk (ASYNCcnt-tracked DMA path)
__device__ __forceinline__ void async_cp16(unsigned lds_off, const void* gptr) {
    asm volatile("global_load_async_to_lds_b128 %0, %1, off"
                 :: "v"(lds_off), "v"((unsigned long)gptr) : "memory");
}
__device__ __forceinline__ void async_wait0() {
    asm volatile("s_wait_asynccnt 0" ::: "memory");
}

#define LDP  40   // LDS row stride (bf16) for 32-wide tiles in k_ktv
#define LDP2 72   // LDS row stride (bf16) for 64-wide GEMM tiles (144B, 16B-aligned chunks)

// =====================================================================
// Kernel 0: fp32 -> bf16 conversion pass (x, Wqkv_w, out_w)
// =====================================================================
__global__ __launch_bounds__(256) void k_f2bf(const float* __restrict__ src,
                                              unsigned short* __restrict__ dst, int n)
{
    int idx = blockIdx.x * 256 + threadIdx.x;
    int i4 = idx * 4;
    if (i4 < n) {
        float4 v = *reinterpret_cast<const float4*>(src + i4);
        uint2 u;
        u.x = (unsigned)f2bf(v.x) | ((unsigned)f2bf(v.y) << 16);
        u.y = (unsigned)f2bf(v.z) | ((unsigned)f2bf(v.w) << 16);
        *reinterpret_cast<uint2*>(dst + i4) = u;
    }
}

// =====================================================================
// Shared bf16 NT GEMM: C(MxN) = A(MxK=1024) @ B(NxK=1024)^T + bias
// block 256 thr (8 waves), tile 128x64, wave 32x32, k-step 64
// A/B staged to LDS with async global->LDS DMA (no VGPR round-trip)
// NSTR = output row stride; F32OUT: write float, else bf16
// =====================================================================
template<int NSTR, bool F32OUT>
__global__ __launch_bounds__(256) void k_gemm(const unsigned short* __restrict__ A,
                                              const unsigned short* __restrict__ Bw,
                                              const float* __restrict__ bias,
                                              unsigned short* __restrict__ outb,
                                              float* __restrict__ outf)
{
    __shared__ unsigned short As[128 * LDP2];
    __shared__ unsigned short Bs[64 * LDP2];
    const int tid = threadIdx.x, lane = tid & 31, wv = tid >> 5;
    const int wm = wv >> 1, wn = wv & 1;
    const int rowBase = blockIdx.y * 128, colBase = blockIdx.x * 64;
    const int hv = lane >> 4, ln = lane & 15;

    v8f acc[2][2];
    #pragma unroll
    for (int i = 0; i < 2; ++i)
        #pragma unroll
        for (int j = 0; j < 2; ++j) acc[i][j] = v8f_zero();

    for (int k0 = 0; k0 < 1024; k0 += 64) {
        // stage A: 128x64 bf16 = 1024 x 16B chunks, 4 per thread (async DMA)
        #pragma unroll
        for (int i = 0; i < 4; ++i) {
            int idx = tid + i * 256; int r = idx >> 3, c8 = (idx & 7) * 8;
            unsigned lds = (unsigned)(uintptr_t)&As[r * LDP2 + c8];
            async_cp16(lds, A + (size_t)(rowBase + r) * 1024 + k0 + c8);
        }
        // stage B: 64x64 bf16 = 512 x 16B chunks, 2 per thread
        #pragma unroll
        for (int i = 0; i < 2; ++i) {
            int idx = tid + i * 256; int r = idx >> 3, c8 = (idx & 7) * 8;
            unsigned lds = (unsigned)(uintptr_t)&Bs[r * LDP2 + c8];
            async_cp16(lds, Bw + (size_t)(colBase + r) * 1024 + k0 + c8);
        }
        if (k0 + 64 < 1024) {
            __builtin_prefetch(A  + (size_t)(rowBase + (tid >> 1)) * 1024 + k0 + 64, 0, 3);
            __builtin_prefetch(Bw + (size_t)(colBase + (tid >> 2)) * 1024 + k0 + 64, 0, 3);
        }
        async_wait0();           // this wave's DMA chunks are in LDS
        __syncthreads();         // all waves' chunks visible

        #pragma unroll
        for (int ks = 0; ks < 2; ++ks) {
            v16bf af[2], bfv[2];
            #pragma unroll
            for (int mi = 0; mi < 2; ++mi) {
                const unsigned short* rp = &As[(wm * 32 + mi * 16 + ln) * LDP2 + ks * 32];
                af[mi] = ld_frag(rp + hv * 8, rp + 16 + hv * 8);
            }
            #pragma unroll
            for (int ni = 0; ni < 2; ++ni) {
                const unsigned short* rp = &Bs[(wn * 32 + ni * 16 + ln) * LDP2 + ks * 32];
                bfv[ni] = ld_frag(rp + hv * 8, rp + 16 + hv * 8);
            }
            #pragma unroll
            for (int mi = 0; mi < 2; ++mi)
                #pragma unroll
                for (int ni = 0; ni < 2; ++ni)
                    acc[mi][ni] = __builtin_amdgcn_wmma_f32_16x16x32_bf16(
                        false, af[mi], false, bfv[ni], (short)0, acc[mi][ni], false, false);
        }
        __syncthreads();
    }

    #pragma unroll
    for (int mi = 0; mi < 2; ++mi)
        #pragma unroll
        for (int ni = 0; ni < 2; ++ni) {
            int gn = colBase + wn * 32 + ni * 16 + ln;
            float bv = bias[gn];
            #pragma unroll
            for (int r = 0; r < 8; ++r) {
                int gm = rowBase + wm * 32 + mi * 16 + r + hv * 8;
                float v = acc[mi][ni][r] + bv;
                if (F32OUT) outf[(size_t)gm * NSTR + gn] = v;
                else        outb[(size_t)gm * NSTR + gn] = f2bf(v);
            }
        }
}

// =====================================================================
// Kernel 2: per (b,h): KtV = K^T @ V (32x32), store transposed bf16
// block = 128 thr (4 waves), each wave covers 512 s, k-step 32
// =====================================================================
__global__ __launch_bounds__(128) void k_ktv(const unsigned short* __restrict__ qkv,
                                             unsigned short* __restrict__ W2)
{
    __shared__ unsigned short Kt[4 * 32 * LDP];
    __shared__ unsigned short Vt[4 * 32 * LDP];
    __shared__ float red[4 * 1024];
    const int tid = threadIdx.x, lane = tid & 31, wv = tid >> 5;
    const int bh = blockIdx.x, b = bh >> 5, h = bh & 31;
    const size_t rowB = (size_t)b * SEQ;
    const int hv = lane >> 4, ln = lane & 15;

    v8f acc[2][2];
    #pragma unroll
    for (int i = 0; i < 2; ++i)
        #pragma unroll
        for (int j = 0; j < 2; ++j) acc[i][j] = v8f_zero();

    for (int it = 0; it < 16; ++it) {
        int s0 = wv * 512 + it * 32;
        // load 32x32 K,V bf16 tiles and transpose into LDS as [d][s]
        #pragma unroll
        for (int p = 0; p < 4; ++p) {
            int sl = p * 8 + (lane >> 2);
            int dc = (lane & 3) * 8;
            size_t base = (rowB + s0 + sl) * NQKV;
            v4u kq = *reinterpret_cast<const v4u*>(qkv + base + 1024 + h * 32 + dc);
            v4u vq = *reinterpret_cast<const v4u*>(qkv + base + 2048 + h * 32 + dc);
            const unsigned short* kp = (const unsigned short*)&kq;
            const unsigned short* vp = (const unsigned short*)&vq;
            #pragma unroll
            for (int j = 0; j < 8; ++j) {
                Kt[(wv * 32 + dc + j) * LDP + sl] = kp[j];
                Vt[(wv * 32 + dc + j) * LDP + sl] = vp[j];
            }
        }
        __syncthreads();
        v16bf af[2], bfv[2];
        #pragma unroll
        for (int mi = 0; mi < 2; ++mi) {
            const unsigned short* rp = &Kt[(wv * 32 + mi * 16 + ln) * LDP];
            af[mi] = ld_frag(rp + hv * 8, rp + 16 + hv * 8);
        }
        #pragma unroll
        for (int ni = 0; ni < 2; ++ni) {
            const unsigned short* rp = &Vt[(wv * 32 + ni * 16 + ln) * LDP];
            bfv[ni] = ld_frag(rp + hv * 8, rp + 16 + hv * 8);
        }
        #pragma unroll
        for (int mi = 0; mi < 2; ++mi)
            #pragma unroll
            for (int ni = 0; ni < 2; ++ni)
                acc[mi][ni] = __builtin_amdgcn_wmma_f32_16x16x32_bf16(
                    false, af[mi], false, bfv[ni], (short)0, acc[mi][ni], false, false);
        __syncthreads();
    }

    // per-wave partials, then fixed-order reduce (deterministic)
    #pragma unroll
    for (int mi = 0; mi < 2; ++mi)
        #pragma unroll
        for (int ni = 0; ni < 2; ++ni)
            #pragma unroll
            for (int r = 0; r < 8; ++r) {
                int m = mi * 16 + r + hv * 8;
                int n = ni * 16 + ln;
                red[wv * 1024 + m * 32 + n] = acc[mi][ni][r];
            }
    __syncthreads();
    for (int i = tid; i < 1024; i += 128) {
        float s = red[i] + red[1024 + i] + red[2048 + i] + red[3072 + i];
        int m = i >> 5, n = i & 31;
        W2[(size_t)bh * 1024 + n * 32 + m] = f2bf(s);   // store (KtV)^T row-major
    }
}

// =====================================================================
// Kernel 3: ctx = Q @ KtV (per b,h), fp32 out + groupnorm partial stats
// grid (S/128, B*H), block 256 (8 waves, 16 rows each)
// =====================================================================
__global__ __launch_bounds__(256) void k_ctx(const unsigned short* __restrict__ qkv,
                                             const unsigned short* __restrict__ W2,
                                             float* __restrict__ ctx,
                                             float* __restrict__ partials)
{
    __shared__ float rs[256], rq[256];
    const int tid = threadIdx.x, lane = tid & 31, wv = tid >> 5;
    const int bh = blockIdx.y, b = bh >> 5, h = bh & 31;
    const int sTile = blockIdx.x * 128;
    const int hv = lane >> 4, ln = lane & 15;

    v16bf bfv[2];
    #pragma unroll
    for (int ni = 0; ni < 2; ++ni) {
        const unsigned short* rp = W2 + (size_t)bh * 1024 + (ni * 16 + ln) * 32;
        bfv[ni] = ld_frag(rp + hv * 8, rp + 16 + hv * 8);
    }
    int srow = sTile + wv * 16 + ln;
    const unsigned short* ap = qkv + (size_t)(b * SEQ + srow) * NQKV + h * 32;
    v16bf af = ld_frag(ap + hv * 8, ap + 16 + hv * 8);

    float ls = 0.f, lsq = 0.f;
    #pragma unroll
    for (int ni = 0; ni < 2; ++ni) {
        v8f c = __builtin_amdgcn_wmma_f32_16x16x32_bf16(
            false, af, false, bfv[ni], (short)0, v8f_zero(), false, false);
        #pragma unroll
        for (int r = 0; r < 8; ++r) {
            int m = r + hv * 8;
            int so = sTile + wv * 16 + m;
            int col = h * 32 + ni * 16 + ln;
            float v = c[r];
            ctx[(size_t)(b * SEQ + so) * D_MODEL + col] = v;
            ls += v; lsq += v * v;
        }
    }
    rs[tid] = ls; rq[tid] = lsq;
    __syncthreads();
    for (int off = 128; off > 0; off >>= 1) {
        if (tid < off) { rs[tid] += rs[tid + off]; rq[tid] += rq[tid + off]; }
        __syncthreads();
    }
    if (tid == 0) {
        partials[(size_t)(bh * 16 + blockIdx.x) * 2 + 0] = rs[0];
        partials[(size_t)(bh * 16 + blockIdx.x) * 2 + 1] = rq[0];
    }
}

// =====================================================================
// Kernel 4: finalize mean/rstd per (b,h)  — deterministic fixed-order sum
// =====================================================================
__global__ void k_stats(const float* __restrict__ partials, float* __restrict__ mr)
{
    int bh = threadIdx.x;
    if (bh < 64) {
        float s = 0.f, q = 0.f;
        for (int i = 0; i < 16; ++i) {
            s += partials[(bh * 16 + i) * 2 + 0];
            q += partials[(bh * 16 + i) * 2 + 1];
        }
        const float inv = 1.0f / 65536.0f;           // Dh * S
        float mean = s * inv;
        float var  = q * inv - mean * mean;
        mr[bh * 2 + 0] = mean;
        mr[bh * 2 + 1] = rsqrtf(var + 1e-5f);
    }
}

// =====================================================================
// Kernel 4.5: groupnorm + affine, fp32 ctx -> bf16 ctxn
// =====================================================================
__global__ __launch_bounds__(256) void k_norm(const float* __restrict__ ctx,
                                              const float* __restrict__ gw,
                                              const float* __restrict__ gb,
                                              const float* __restrict__ mr,
                                              unsigned short* __restrict__ ctxn)
{
    int idx = blockIdx.x * 256 + threadIdx.x;        // over 4096*1024/4 float4s
    int m = idx >> 8;                                 // 256 float4 per row
    int c4 = (idx & 255) * 4;
    int b = m >> 11, h = c4 >> 5;
    float mean = mr[(b * 32 + h) * 2 + 0];
    float rstd = mr[(b * 32 + h) * 2 + 1];
    float4 v  = *reinterpret_cast<const float4*>(ctx + (size_t)m * 1024 + c4);
    float4 w4 = *reinterpret_cast<const float4*>(gw + c4);
    float4 g4 = *reinterpret_cast<const float4*>(gb + c4);
    uint2 u;
    u.x = (unsigned)f2bf((v.x - mean) * rstd * w4.x + g4.x)
        | ((unsigned)f2bf((v.y - mean) * rstd * w4.y + g4.y) << 16);
    u.y = (unsigned)f2bf((v.z - mean) * rstd * w4.z + g4.z)
        | ((unsigned)f2bf((v.w - mean) * rstd * w4.w + g4.w) << 16);
    *reinterpret_cast<uint2*>(ctxn + (size_t)m * 1024 + c4) = u;
}

// =====================================================================
extern "C" void kernel_launch(void* const* d_in, const int* in_sizes, int n_in,
                              void* d_out, int out_size, void* d_ws, size_t ws_size,
                              hipStream_t stream)
{
    (void)in_sizes; (void)n_in; (void)out_size; (void)ws_size;
    const float* x      = (const float*)d_in[0];
    const float* Wqkv_w = (const float*)d_in[1];
    const float* Wqkv_b = (const float*)d_in[2];
    const float* gn_w   = (const float*)d_in[3];
    const float* gn_b   = (const float*)d_in[4];
    const float* out_w  = (const float*)d_in[5];
    const float* out_b  = (const float*)d_in[6];
    float* out = (float*)d_out;

    char* ws = (char*)d_ws;
    unsigned short* qkv  = (unsigned short*)(ws);              // 25,165,824 B
    float*          ctx  = (float*)(ws + 25165824);            // 16,777,216 B
    unsigned short* W2   = (unsigned short*)(ws + 41943040);   //    131,072 B
    float*     partials  = (float*)(ws + 42074112);            //      8,192 B
    float*          mr   = (float*)(ws + 42082304);            //        512 B
    unsigned short* xb   = (unsigned short*)(ws + 42082816);   //  8,388,608 B
    unsigned short* wqb  = (unsigned short*)(ws + 50471424);   //  6,291,456 B
    unsigned short* owb  = (unsigned short*)(ws + 56762880);   //  2,097,152 B
    unsigned short* ctxn = (unsigned short*)(ws + 58860032);   //  8,388,608 B

    // 0) one-time (per launch) fp32->bf16 conversions
    k_f2bf<<<dim3((MROWS * D_MODEL) / 1024), 256, 0, stream>>>(x, xb, MROWS * D_MODEL);
    k_f2bf<<<dim3((NQKV * D_MODEL) / 1024), 256, 0, stream>>>(Wqkv_w, wqb, NQKV * D_MODEL);
    k_f2bf<<<dim3((D_MODEL * D_MODEL) / 1024), 256, 0, stream>>>(out_w, owb, D_MODEL * D_MODEL);

    // 1) qkv = x @ Wqkv^T + b  (bf16 out)
    k_gemm<NQKV, false><<<dim3(NQKV / 64, MROWS / 128), 256, 0, stream>>>(
        xb, wqb, Wqkv_b, qkv, nullptr);
    // 2) per-(b,h) KtV^T
    k_ktv<<<dim3(BATCH * N_HEADS), 128, 0, stream>>>(qkv, W2);
    // 3) ctx = Q @ KtV + stats partials
    k_ctx<<<dim3(SEQ / 128, BATCH * N_HEADS), 256, 0, stream>>>(qkv, W2, ctx, partials);
    // 4) mean / rstd
    k_stats<<<dim3(1), 64, 0, stream>>>(partials, mr);
    // 4.5) groupnorm -> bf16
    k_norm<<<dim3((MROWS * D_MODEL) / 1024), 256, 0, stream>>>(ctx, gn_w, gn_b, mr, ctxn);
    // 5) out = ctxn @ out_w^T + out_b  (fp32 out)
    k_gemm<D_MODEL, true><<<dim3(D_MODEL / 64, MROWS / 128), 256, 0, stream>>>(
        ctxn, owb, out_b, nullptr, out);
}
